// Network_58196806861387
// MI455X (gfx1250) — matrix operand order repaired and checked by reading, not compiled, not run on hardware
//
#include <hip/hip_runtime.h>
#include <cstddef>

typedef __attribute__((ext_vector_type(16))) __bf16 v16bf;
typedef __attribute__((ext_vector_type(8)))  __bf16 v8bf;
typedef __attribute__((ext_vector_type(8)))  float  v8f;

#define OMEGA 2.0f
#define MID   256

// ---------------------------------------------------------------------------
// SIREN: h1 = sin(w*(x*w1+b1)); h2 = sin(w*(h1@w2.T+b2)); out = h2@w3.T+b3
// one block per row, 256 threads. out_bf (U/V nets) or out_f (W net).
// ---------------------------------------------------------------------------
__global__ void siren_kernel(const float* __restrict__ x,
                             const float* __restrict__ w1, const float* __restrict__ b1,
                             const float* __restrict__ w2, const float* __restrict__ b2,
                             const float* __restrict__ w3, const float* __restrict__ b3,
                             int r_out, __bf16* out_bf, float* out_f) {
    __shared__ float h1[MID];
    __shared__ float h2[MID];
    const int row = blockIdx.x;
    const int c   = threadIdx.x;

    float xv = x[row];
    h1[c] = sinf(OMEGA * (xv * w1[c] + b1[c]));
    __syncthreads();

    float acc = b2[c];
    const float* w2r = w2 + (size_t)c * MID;
    #pragma unroll 8
    for (int d = 0; d < MID; ++d) acc = fmaf(h1[d], w2r[d], acc);
    float h2v = sinf(OMEGA * acc);
    h2[c] = h2v;
    __syncthreads();

    if (c < r_out) {
        float o = b3[c];
        const float* w3r = w3 + (size_t)c * MID;
        #pragma unroll 8
        for (int d = 0; d < MID; ++d) o = fmaf(h2[d], w3r[d], o);
        if (out_bf) out_bf[(size_t)row * r_out + c] = (__bf16)o;
        else        out_f [(size_t)row * r_out + c] = o;
    }
}

// ---------------------------------------------------------------------------
// conv1: img [512,512,3] HWC -> y1 [16,256,256], 3x3, stride 2, pad 1
// grid (oh=256, co=16), block 256 (ow)
// ---------------------------------------------------------------------------
__global__ void conv1_kernel(const float* __restrict__ img,
                             const float* __restrict__ c1w,
                             const float* __restrict__ c1b,
                             float* __restrict__ y1) {
    __shared__ float w[27];
    const int ow = threadIdx.x;
    const int oh = blockIdx.x;
    const int co = blockIdx.y;
    if (threadIdx.x < 27) w[threadIdx.x] = c1w[co * 27 + threadIdx.x];
    __syncthreads();

    float acc = c1b[co];
    #pragma unroll
    for (int dy = 0; dy < 3; ++dy) {
        int h = 2 * oh + dy - 1;
        if (h < 0 || h >= 512) continue;
        #pragma unroll
        for (int dx = 0; dx < 3; ++dx) {
            int ww = 2 * ow + dx - 1;
            if (ww < 0 || ww >= 512) continue;
            const float* p = img + ((size_t)h * 512 + ww) * 3;
            #pragma unroll
            for (int ci = 0; ci < 3; ++ci)
                acc = fmaf(p[ci], w[ci * 9 + dy * 3 + dx], acc);
        }
    }
    y1[((size_t)co * 256 + oh) * 256 + ow] = acc;
}

// ---------------------------------------------------------------------------
// per-channel sum / sumsq over 256*256 values. grid 16, block 256. no atomics.
// ---------------------------------------------------------------------------
__global__ void stats_kernel(const float* __restrict__ y1, float* __restrict__ gstats) {
    __shared__ float rs[256];
    __shared__ float rq[256];
    const int c = blockIdx.x;
    const int tid = threadIdx.x;
    const float* p = y1 + (size_t)c * 65536;
    float s = 0.f, sq = 0.f;
    for (int idx = tid; idx < 65536; idx += 256) {
        float v = p[idx];
        s += v;
        sq = fmaf(v, v, sq);
    }
    rs[tid] = s; rq[tid] = sq;
    __syncthreads();
    for (int off = 128; off > 0; off >>= 1) {
        if (tid < off) { rs[tid] += rs[tid + off]; rq[tid] += rq[tid + off]; }
        __syncthreads();
    }
    if (tid == 0) { gstats[2 * c] = rs[0]; gstats[2 * c + 1] = rq[0]; }
}

// ---------------------------------------------------------------------------
// BN(training stats) + tanh, written into a zero-padded [16][258][258] buffer
// ---------------------------------------------------------------------------
__global__ void bn_tanh_pad_kernel(const float* __restrict__ y1,
                                   const float* __restrict__ gstats,
                                   const float* __restrict__ g,
                                   const float* __restrict__ b,
                                   float* __restrict__ yt) {
    const int total = 16 * 258 * 258;
    int idx = blockIdx.x * 256 + threadIdx.x;
    if (idx >= total) return;
    int c   = idx / (258 * 258);
    int rem = idx % (258 * 258);
    int hh  = rem / 258;
    int ww  = rem % 258;
    float v = 0.f;
    if (hh >= 1 && hh <= 256 && ww >= 1 && ww <= 256) {
        float mean = gstats[2 * c] * (1.f / 65536.f);
        float var  = gstats[2 * c + 1] * (1.f / 65536.f) - mean * mean;
        float rstd = rsqrtf(var + 1e-5f);
        float y = y1[((size_t)c * 256 + (hh - 1)) * 256 + (ww - 1)];
        v = tanhf((y - mean) * rstd * g[c] + b[c]);
    }
    yt[idx] = v;
}

// ---------------------------------------------------------------------------
// conv2 (32<-16ch, 3x3, s1, p1 on padded yt) fused with column (row-index h)
// sum: s[w][co] = sum_h conv2(co,h,w).  grid (w=256, co=32), block 256 (h).
// ---------------------------------------------------------------------------
__global__ void conv2_colsum_kernel(const float* __restrict__ yt,
                                    const float* __restrict__ c2w,
                                    const float* __restrict__ c2b,
                                    float* __restrict__ s) {
    __shared__ float wt[144];
    __shared__ float red[256];
    const int h  = threadIdx.x;
    const int w  = blockIdx.x;
    const int co = blockIdx.y;
    if (h < 144) wt[h] = c2w[co * 144 + h];
    __syncthreads();

    float acc = c2b[co];
    #pragma unroll 4
    for (int ci = 0; ci < 16; ++ci) {
        const float* base = yt + ((size_t)ci * 258 + h) * 258 + w;
        #pragma unroll
        for (int dy = 0; dy < 3; ++dy)
            #pragma unroll
            for (int dx = 0; dx < 3; ++dx)
                acc = fmaf(base[dy * 258 + dx], wt[ci * 9 + dy * 3 + dx], acc);
    }
    red[h] = acc;
    __syncthreads();
    for (int off = 128; off > 0; off >>= 1) {
        if (h < off) red[h] += red[h + off];
        __syncthreads();
    }
    if (h == 0) s[(size_t)w * 32 + co] = red[0];
}

// ---------------------------------------------------------------------------
// Tt[k][b] = sum_c s[b][c] * W[k][c]   (T transposed, bf16 for WMMA B operand)
// grid 256 (k), block 256 (b)
// ---------------------------------------------------------------------------
__global__ void tmat_kernel(const float* __restrict__ s,
                            const float* __restrict__ Wf,
                            __bf16* __restrict__ Tt) {
    __shared__ float wk[32];
    const int b = threadIdx.x;
    const int k = blockIdx.x;
    if (b < 32) wk[b] = Wf[(size_t)k * 32 + b];
    __syncthreads();
    float acc = 0.f;
    const float* sr = s + (size_t)b * 32;
    #pragma unroll
    for (int c = 0; c < 32; ++c) acc = fmaf(sr[c], wk[c], acc);
    Tt[(size_t)k * 256 + b] = (__bf16)acc;
}

// ---------------------------------------------------------------------------
// out[i,j,k] = sum_b U[i,b]*V[j,b]*T[b,k]  via v_wmma_f32_16x16x32_bf16.
// Register-blocked: one wave = (j0 tile, 8 consecutive i, 2 k-tiles).
// Per bb-step: V frag + 2 B frags loaded ONCE, reused by 8 A frags (one per i)
// -> 8x less L2 traffic on the B operand than one-tile-per-wave.
// 128 WMMAs / wave; acc = 8i x 2k x v8f = 128 VGPRs.
// grid (j0=32, ig=64, kg=8), block 32 (one wave; EXEC all-ones, no divergence).
// ---------------------------------------------------------------------------
__global__ void __launch_bounds__(32)
einsum_wmma_kernel(const __bf16* __restrict__ Ubf,
                   const __bf16* __restrict__ Vbf,
                   const __bf16* __restrict__ Tt,
                   float* __restrict__ out) {
    const int lane = threadIdx.x;
    const int j0   = blockIdx.x;      // 0..31  (j tile)
    const int ig   = blockIdx.y;      // 0..63  (i group of 8)
    const int kg   = blockIdx.z;      // 0..7   (k-tile pair)
    const int m    = lane & 15;
    const int hi   = (lane >> 4) & 1;
    const int kbA  = hi ? 8 : 0;      // 16-bit A 16x32: lanes<16 K in {0..7,16..23}
    const int kbB  = hi ? 16 : 0;     // B: lanes<16 K in {0..7,16..23} like A (kbB cancelled at load)
    const int j    = j0 * 16 + m;
    const int n    = m;               // output column within k tile

    const __bf16* vrow  = Vbf + (size_t)j * 256;
    const __bf16* ubase = Ubf + (size_t)(ig * 8) * 256;
    const __bf16* t0row = Tt + (size_t)((kg * 2 + 0) * 16 + n) * 256 + kbB;
    const __bf16* t1row = Tt + (size_t)((kg * 2 + 1) * 16 + n) * 256 + kbB;

    v8f acc[8][2] = {};

    #pragma unroll
    for (int bb = 0; bb < 8; ++bb) {
        const int b0 = bb * 32;
        v8bf v0 = *(const v8bf*)(vrow + b0 + kbA);        // K = b0+kbA .. +7
        v8bf v1 = *(const v8bf*)(vrow + b0 + kbA + 16);   // K = b0+kbA+16 .. +23
        v8bf t00 = *(const v8bf*)(t0row - kbB + b0 + kbA);       // K = b0+kbA .. +7
        v8bf t01 = *(const v8bf*)(t0row - kbB + b0 + kbA + 16);  // K = b0+kbA+16 .. +23
        v8bf t10 = *(const v8bf*)(t1row - kbB + b0 + kbA);
        v8bf t11 = *(const v8bf*)(t1row - kbB + b0 + kbA + 16);
        v16bf bf0 = __builtin_shufflevector(t00, t01, 0,1,2,3,4,5,6,7,8,9,10,11,12,13,14,15);
        v16bf bf1 = __builtin_shufflevector(t10, t11, 0,1,2,3,4,5,6,7,8,9,10,11,12,13,14,15);
        #pragma unroll
        for (int ii = 0; ii < 8; ++ii) {
            v8bf u0 = *(const v8bf*)(ubase + (size_t)ii * 256 + b0 + kbA);
            v8bf u1 = *(const v8bf*)(ubase + (size_t)ii * 256 + b0 + kbA + 16);
            v16bf a;
            #pragma unroll
            for (int q = 0; q < 8; ++q) {
                a[q]     = (__bf16)((float)u0[q] * (float)v0[q]);
                a[q + 8] = (__bf16)((float)u1[q] * (float)v1[q]);
            }
            acc[ii][0] = __builtin_amdgcn_wmma_f32_16x16x32_bf16(
                false, a, false, bf0, (short)0, acc[ii][0], false, false);
            acc[ii][1] = __builtin_amdgcn_wmma_f32_16x16x32_bf16(
                false, a, false, bf1, (short)0, acc[ii][1], false, false);
        }
    }

    // C/D layout: lane -> col n; VGPR r -> row (hi*8 + r)
    #pragma unroll
    for (int ii = 0; ii < 8; ++ii) {
        const size_t rowbase = (size_t)((ig * 8 + ii) * 512 + j0 * 16 + hi * 8);
        #pragma unroll
        for (int kt = 0; kt < 2; ++kt) {
            float* po = out + rowbase * 256 + (size_t)(kg * 2 + kt) * 16 + n;
            #pragma unroll
            for (int r = 0; r < 8; ++r) po[(size_t)r * 256] = acc[ii][kt][r];
        }
    }
}

// ---------------------------------------------------------------------------
extern "C" void kernel_launch(void* const* d_in, const int* in_sizes, int n_in,
                              void* d_out, int out_size, void* d_ws, size_t ws_size,
                              hipStream_t stream) {
    const float* img = (const float*)d_in[0];
    const float* Uin = (const float*)d_in[1];
    const float* Vin = (const float*)d_in[2];
    const float* Win = (const float*)d_in[3];
    const float* Uw[6] = {(const float*)d_in[4],  (const float*)d_in[5],  (const float*)d_in[6],
                          (const float*)d_in[7],  (const float*)d_in[8],  (const float*)d_in[9]};
    const float* Vw[6] = {(const float*)d_in[10], (const float*)d_in[11], (const float*)d_in[12],
                          (const float*)d_in[13], (const float*)d_in[14], (const float*)d_in[15]};
    const float* Ww[6] = {(const float*)d_in[16], (const float*)d_in[17], (const float*)d_in[18],
                          (const float*)d_in[19], (const float*)d_in[20], (const float*)d_in[21]};
    const float* c1w = (const float*)d_in[22];
    const float* c1b = (const float*)d_in[23];
    const float* bng = (const float*)d_in[24];
    const float* bnb = (const float*)d_in[25];
    const float* c2w = (const float*)d_in[26];
    const float* c2b = (const float*)d_in[27];
    float* out = (float*)d_out;

    // workspace layout (all offsets 256B aligned); total ~8.8 MB
    char* ws = (char*)d_ws;
    __bf16* Ubf   = (__bf16*)(ws + 0);              // 512*256*2  = 262144
    __bf16* Vbf   = (__bf16*)(ws + 262144);         // 262144
    float*  Wf    = (float*)(ws + 524288);          // 256*32*4   = 32768
    float*  y1    = (float*)(ws + 557056);          // 16*256*256*4 = 4194304
    float*  gst   = (float*)(ws + 4751360);         // 32*4 (pad 256)
    float*  yt    = (float*)(ws + 4751616);         // 16*258*258*4 = 4260096
    float*  smat  = (float*)(ws + 9011712);         // 256*32*4   = 32768
    __bf16* Tt    = (__bf16*)(ws + 9044480);        // 256*256*2  = 131072
                                                    // end: 9175552 bytes

    // SIREN nets
    siren_kernel<<<512, 256, 0, stream>>>(Uin, Uw[0], Uw[1], Uw[2], Uw[3], Uw[4], Uw[5],
                                          256, Ubf, nullptr);
    siren_kernel<<<512, 256, 0, stream>>>(Vin, Vw[0], Vw[1], Vw[2], Vw[3], Vw[4], Vw[5],
                                          256, Vbf, nullptr);
    siren_kernel<<<256, 256, 0, stream>>>(Win, Ww[0], Ww[1], Ww[2], Ww[3], Ww[4], Ww[5],
                                          32, nullptr, Wf);

    // conv path
    conv1_kernel<<<dim3(256, 16), 256, 0, stream>>>(img, c1w, c1b, y1);
    stats_kernel<<<16, 256, 0, stream>>>(y1, gst);
    {
        int total = 16 * 258 * 258;
        bn_tanh_pad_kernel<<<(total + 255) / 256, 256, 0, stream>>>(y1, gst, bng, bnb, yt);
    }
    conv2_colsum_kernel<<<dim3(256, 32), 256, 0, stream>>>(yt, c2w, c2b, smat);

    // T = s @ W.T (stored transposed, bf16)
    tmat_kernel<<<256, 256, 0, stream>>>(smat, Wf, Tt);

    // the big einsum on WMMA (register-blocked 8i x 2k per wave)
    einsum_wmma_kernel<<<dim3(32, 64, 8), 32, 0, stream>>>(Ubf, Vbf, Tt, out);

    (void)in_sizes; (void)n_in; (void)out_size; (void)ws_size;
}
